// Transformer_28114855920348
// MI455X (gfx1250) — matrix operand order repaired and checked by reading, not compiled, not run on hardware
//
#include <hip/hip_runtime.h>
#include <math.h>

// ---------------------------------------------------------------------------
// CDNA5 (gfx1250) wave32 WMMA transformer forward.
// Matrix math: v_wmma_f32_16x16x32_bf16, bf16 tiles staged in LDS with
// clause-batched b128 global loads and b64/b128 LDS stores; fragments are
// fetched as contiguous ds_load_b128 pairs per ISA 7.12.2 layouts.
// ---------------------------------------------------------------------------

typedef __attribute__((ext_vector_type(16))) __bf16 v16bf;
typedef __attribute__((ext_vector_type(8)))  __bf16 v8bf;
typedef __attribute__((ext_vector_type(4)))  __bf16 v4bf;
typedef __attribute__((ext_vector_type(8)))  float  v8f;

#define LAYERS 6
#define NH     8
#define DM     512
#define DKH    64
#define DFFN   2048
#define NB     8
#define SEQ    512
#define LN_EPS 1e-5f

// ---------------- wave32 reductions ----------------
__device__ __forceinline__ float wave_sum32(float v) {
#pragma unroll
  for (int m = 16; m >= 1; m >>= 1) v += __shfl_xor(v, m, 32);
  return v;
}
__device__ __forceinline__ float wave_max32(float v) {
#pragma unroll
  for (int m = 16; m >= 1; m >>= 1) v = fmaxf(v, __shfl_xor(v, m, 32));
  return v;
}

__device__ __forceinline__ v16bf concat8(v8bf lo, v8bf hi) {
  return __builtin_shufflevector(lo, hi, 0, 1, 2, 3, 4, 5, 6, 7,
                                 8, 9, 10, 11, 12, 13, 14, 15);
}
__device__ __forceinline__ v4bf cvt4(float4 f) {
  v4bf p;
  p[0] = (__bf16)f.x; p[1] = (__bf16)f.y; p[2] = (__bf16)f.z; p[3] = (__bf16)f.w;
  return p;
}

// ---------------------------------------------------------------------------
// Universal batched GEMM:  C = act( alpha * A @ op(B) + bias )
//   A: M x K row-major (lda), op(B): K x N (ldb; TRANSB => B stored N x K),
//   C: M x N (ldc). Batch index z = z1*Z2 + z2 with independent strides.
// Block: 256 threads = 8 waves. Tile 128(M) x 64(N), K step 32.
// Wave grid 4(M) x 2(N): each wave owns 32x32 -> 4 WMMA accumulators fed by
// 2 A fragments + 2 B fragments (each two ds_load_b128's).
// Staging: load ALL float4s of the k-slice into distinct registers first
// (one VMEM clause, one wait), then convert+store to LDS.
// ---------------------------------------------------------------------------
template <int TRANSB>
__global__ void __launch_bounds__(256)
gemm_bf16_wmma(const float* __restrict__ A, const float* __restrict__ Bm,
               const float* __restrict__ bias, float* __restrict__ C,
               int K, int lda, int ldb, int ldc,
               float alpha, int relu, int Z2,
               long long sA1, long long sA2, long long sB1, long long sB2,
               long long sC1, long long sC2)
{
  __shared__ __bf16 As[128][32];   // M-major, K contiguous (8 KB)
  __shared__ __bf16 BsT[64][32];   // N-major, K contiguous (4 KB)

  const int z  = blockIdx.z;
  const int z1 = z / Z2, z2 = z - z1 * Z2;
  A  += (long long)z1 * sA1 + (long long)z2 * sA2;
  Bm += (long long)z1 * sB1 + (long long)z2 * sB2;
  C  += (long long)z1 * sC1 + (long long)z2 * sC2;

  const int tid    = threadIdx.x;
  const int lane   = tid & 31;
  const int wave   = tid >> 5;
  const int mOff   = (wave & 3) * 32;   // 4 waves tile M (32 rows each)
  const int nOff   = (wave >> 2) * 32;  // 2 waves tile N (32 cols each)
  const int blockM = blockIdx.y * 128;
  const int blockN = blockIdx.x * 64;

  v8f acc00 = {}, acc01 = {}, acc10 = {}, acc11 = {};

  // ISA 7.12.2 fragment geometry (wave32)
  const int kbA = (lane & 16) ? 8 : 0;    // A: half-wave K offset
  const int kbB = (lane & 16) ? 8 : 0;    // B: half-wave K offset
  const int mA0 = mOff + (lane & 15);
  const int mA1 = mA0 + 16;
  const int nB0 = nOff + (lane & 15);
  const int nB1 = nB0 + 16;

  // staging geometry
  const int rQ  = tid >> 3;        // 0..31 (row step 32 per iteration)
  const int cQ  = (tid & 7) * 4;   // 0..28, quad of K
  const int nS  = tid & 63;        // B (no-trans): fixed n per thread
  const int kS  = (tid >> 6) * 8;  // B (no-trans): 8-wide K run

  // per-thread base pointers (advance by 32 floats per k-step)
  const float* aP = A + (size_t)(blockM + rQ) * lda + cQ;
  const float* bP = TRANSB ? (Bm + (size_t)(blockN + rQ) * ldb + cQ)
                           : (Bm + (size_t)kS * ldb + (blockN + nS));

  for (int k0 = 0; k0 < K; k0 += 32) {
    // ---- phase 1: issue ALL global loads of this k-slice (one clause) ----
    float4 fa0 = *(const float4*)(aP + (size_t)0   * lda);
    float4 fa1 = *(const float4*)(aP + (size_t)32  * lda);
    float4 fa2 = *(const float4*)(aP + (size_t)64  * lda);
    float4 fa3 = *(const float4*)(aP + (size_t)96  * lda);
    float4 fb0, fb1;
    float  fs[8];
    if (TRANSB) {
      fb0 = *(const float4*)(bP);
      fb1 = *(const float4*)(bP + (size_t)32 * ldb);
    } else {
#pragma unroll
      for (int j = 0; j < 8; ++j) fs[j] = bP[(size_t)j * ldb];
    }

    // ---- prefetch next K-slice lines while this slice converts/computes --
    if (k0 + 32 < K) {
      __builtin_prefetch(aP + 32, 0, 3);
      if (TRANSB) __builtin_prefetch(bP + 32, 0, 3);
      else        __builtin_prefetch(bP + (size_t)32 * ldb, 0, 3);
    }

    // ---- phase 2: convert + LDS stores ----
    *(v4bf*)&As[rQ     ][cQ] = cvt4(fa0);
    *(v4bf*)&As[rQ + 32][cQ] = cvt4(fa1);
    *(v4bf*)&As[rQ + 64][cQ] = cvt4(fa2);
    *(v4bf*)&As[rQ + 96][cQ] = cvt4(fa3);
    if (TRANSB) {
      *(v4bf*)&BsT[rQ     ][cQ] = cvt4(fb0);
      *(v4bf*)&BsT[rQ + 32][cQ] = cvt4(fb1);
    } else {
      v8bf p;
#pragma unroll
      for (int j = 0; j < 8; ++j) p[j] = (__bf16)fs[j];
      *(v8bf*)&BsT[nS][kS] = p;
    }
    __syncthreads();

    // ---- fragments: contiguous ds_load_b128 pairs ----
    const v16bf a0 = concat8(*(const v8bf*)&As[mA0][kbA],
                             *(const v8bf*)&As[mA0][16 + kbA]);
    const v16bf a1 = concat8(*(const v8bf*)&As[mA1][kbA],
                             *(const v8bf*)&As[mA1][16 + kbA]);
    const v16bf b0 = concat8(*(const v8bf*)&BsT[nB0][kbB],
                             *(const v8bf*)&BsT[nB0][16 + kbB]);
    const v16bf b1 = concat8(*(const v8bf*)&BsT[nB1][kbB],
                             *(const v8bf*)&BsT[nB1][16 + kbB]);

    acc00 = __builtin_amdgcn_wmma_f32_16x16x32_bf16(false, a0, false, b0,
                                                    (short)0, acc00, false, false);
    acc01 = __builtin_amdgcn_wmma_f32_16x16x32_bf16(false, a0, false, b1,
                                                    (short)0, acc01, false, false);
    acc10 = __builtin_amdgcn_wmma_f32_16x16x32_bf16(false, a1, false, b0,
                                                    (short)0, acc10, false, false);
    acc11 = __builtin_amdgcn_wmma_f32_16x16x32_bf16(false, a1, false, b1,
                                                    (short)0, acc11, false, false);
    __syncthreads();

    aP += 32;
    bP += TRANSB ? 32 : (size_t)32 * ldb;
  }

  // ---- epilogue: D layout VGPR r -> M = r (+8 for upper half-wave) ----
  const int rHalf = (lane & 16) ? 8 : 0;
  const int gn0   = blockN + nOff + (lane & 15);
  const int gn1   = gn0 + 16;
  const float bias0 = bias ? bias[gn0] : 0.f;
  const float bias1 = bias ? bias[gn1] : 0.f;
#pragma unroll
  for (int mi = 0; mi < 2; ++mi) {
    const v8f aN0 = mi ? acc10 : acc00;
    const v8f aN1 = mi ? acc11 : acc01;
    const int rbase = blockM + mOff + mi * 16 + rHalf;
#pragma unroll
    for (int r = 0; r < 8; ++r) {
      float v0 = aN0[r] * alpha + bias0;
      float v1 = aN1[r] * alpha + bias1;
      if (relu) { v0 = fmaxf(v0, 0.f); v1 = fmaxf(v1, 0.f); }
      C[(size_t)(rbase + r) * ldc + gn0] = v0;
      C[(size_t)(rbase + r) * ldc + gn1] = v1;
    }
  }
}

// ---------------- softmax over rows of length SEQ=512 (1 wave / row) -------
__global__ void __launch_bounds__(256)
softmax512(float* __restrict__ p)
{
  const int lane = threadIdx.x & 31;
  const int wave = threadIdx.x >> 5;
  float* row = p + ((size_t)blockIdx.x * 8 + wave) * SEQ;
  float v[16];
  float mx = -3.402823466e38f;
#pragma unroll
  for (int i = 0; i < 16; ++i) { v[i] = row[lane + i * 32]; mx = fmaxf(mx, v[i]); }
  mx = wave_max32(mx);
  float s = 0.f;
#pragma unroll
  for (int i = 0; i < 16; ++i) { v[i] = __expf(v[i] - mx); s += v[i]; }
  s = wave_sum32(s);
  const float inv = 1.f / s;
#pragma unroll
  for (int i = 0; i < 16; ++i) row[lane + i * 32] = v[i] * inv;
}

// --------- LayerNorm over D=512, optional residual add (1 wave / row) ------
__global__ void __launch_bounds__(256)
ln512(const float* __restrict__ x, const float* __restrict__ delta,
      const float* __restrict__ g, const float* __restrict__ b,
      float* __restrict__ out)
{
  const int lane = threadIdx.x & 31;
  const int wave = threadIdx.x >> 5;
  const size_t row = (size_t)blockIdx.x * 8 + wave;
  const float* xr = x + row * DM;
  const float* dr = delta ? (delta + row * DM) : nullptr;
  float v[16];
  float s = 0.f;
#pragma unroll
  for (int i = 0; i < 16; ++i) {
    int c = lane + i * 32;
    v[i] = xr[c] + (dr ? dr[c] : 0.f);
    s += v[i];
  }
  s = wave_sum32(s);
  const float mean = s * (1.f / (float)DM);
  float q = 0.f;
#pragma unroll
  for (int i = 0; i < 16; ++i) { float d = v[i] - mean; q += d * d; }
  q = wave_sum32(q);
  const float rstd = rsqrtf(q * (1.f / (float)DM) + LN_EPS);
  float* orow = out + row * DM;
#pragma unroll
  for (int i = 0; i < 16; ++i) {
    int c = lane + i * 32;
    orow[c] = (v[i] - mean) * rstd * g[c] + b[c];
  }
}

// --------- positional encoding add (matches reference's buggy-arg PE) ------
__global__ void __launch_bounds__(256)
add_pe(const float* __restrict__ in, float* __restrict__ out)
{
  const int idx = blockIdx.x * 256 + threadIdx.x;   // covers NB*SEQ*DM
  const int d = idx & (DM - 1);
  const int s = (idx >> 9) & (SEQ - 1);
  const int i = d >> 1;
  const float inv = __powf(10000.f, (2.f * (float)i) / (float)DM);
  const float t = (float)s;
  const float pe = (d & 1) ? __cosf((t + 1.f) / inv) : __sinf(t / inv);
  out[idx] = in[idx] + pe;
}

// ---------------------------------------------------------------------------
extern "C" void kernel_launch(void* const* d_in, const int* in_sizes, int n_in,
                              void* d_out, int out_size, void* d_ws, size_t ws_size,
                              hipStream_t stream)
{
  (void)in_sizes; (void)n_in; (void)out_size; (void)ws_size;
#define IN(i) ((const float*)d_in[i])
  const float* src      = IN(0);
  const float* tgt      = IN(1);
  const float* enc_Wq   = IN(2);  const float* enc_Wk  = IN(3);
  const float* enc_Wv   = IN(4);  const float* enc_Wo  = IN(5);
  const float* enc_W1   = IN(6);  const float* enc_fb1 = IN(7);
  const float* enc_W2   = IN(8);  const float* enc_fb2 = IN(9);
  const float* enc_ln1g = IN(10); const float* enc_ln1b = IN(11);
  const float* enc_ln2g = IN(12); const float* enc_ln2b = IN(13);
  const float* dec_sWq  = IN(14); const float* dec_sWk = IN(15);
  const float* dec_sWv  = IN(16); const float* dec_sWo = IN(17);
  const float* dec_cWq  = IN(18); const float* dec_cWk = IN(19);
  const float* dec_cWv  = IN(20); const float* dec_cWo = IN(21);
  const float* dec_W1   = IN(22); const float* dec_fb1 = IN(23);
  const float* dec_W2   = IN(24); const float* dec_fb2 = IN(25);
  const float* dec_ln1g = IN(26); const float* dec_ln1b = IN(27);
  const float* dec_ln2g = IN(28); const float* dec_ln2b = IN(29);
  const float* dec_ln3g = IN(30); const float* dec_ln3b = IN(31);
  const float* fin_ln1g = IN(32); const float* fin_ln1b = IN(33);
  const float* fin_ln2g = IN(34); const float* fin_ln2b = IN(35);
#undef IN

  // ---------------- workspace layout (floats) ----------------
  float* ws = (float*)d_ws;
  const size_t NTOK = (size_t)NB * SEQ;   // 4096
  const size_t ACT  = NTOK * DM;          // 2,097,152 (== B*H*S*DK as well)
  size_t off = 0;
  float* mem  = ws + off; off += ACT;
  float* xbuf = ws + off; off += ACT;
  float* qb   = ws + off; off += ACT;
  float* kb   = ws + off; off += ACT;
  float* vb   = ws + off; off += ACT;
  float* pb   = ws + off; off += (size_t)NB * NH * SEQ * SEQ;  // 16,777,216
  float* ocat = ws + off; off += ACT;
  float* t1   = ws + off; off += ACT;
  float* ffb  = ws + off; off += NTOK * DFFN;                  // 8,388,608

  // ---------------- launch helpers ----------------
  auto gemm = [&](const float* A, const float* Bm, const float* bias, float* C,
                  int M, int N, int K, int lda, int ldb, int ldc,
                  int transB, float alpha, int relu, int Z1, int Z2,
                  long long sA1, long long sA2, long long sB1, long long sB2,
                  long long sC1, long long sC2) {
    dim3 grid((unsigned)(N / 64), (unsigned)(M / 128), (unsigned)(Z1 * Z2));
    if (transB)
      gemm_bf16_wmma<1><<<grid, 256, 0, stream>>>(A, Bm, bias, C, K, lda, ldb, ldc,
                                                  alpha, relu, Z2,
                                                  sA1, sA2, sB1, sB2, sC1, sC2);
    else
      gemm_bf16_wmma<0><<<grid, 256, 0, stream>>>(A, Bm, bias, C, K, lda, ldb, ldc,
                                                  alpha, relu, Z2,
                                                  sA1, sA2, sB1, sB2, sC1, sC2);
  };
  auto layernorm = [&](const float* x, const float* delta, const float* g,
                       const float* b, float* outp) {
    ln512<<<dim3((unsigned)(NTOK / 8)), 256, 0, stream>>>(x, delta, g, b, outp);
  };

  const long long WQS = (long long)DM * DKH;   // per-head Wq/Wk/Wv
  const long long QS2 = (long long)SEQ * DKH;  // per-(b,h) Q/K/V
  const long long QS1 = (long long)NH * QS2;
  const long long PS2 = (long long)SEQ * SEQ;  // per-(b,h) scores
  const long long PS1 = (long long)NH * PS2;

  // Multi-head attention: X = queries, KV = keys/values source; out -> delta
  auto attention = [&](const float* X, const float* KV,
                       const float* Wq, const float* Wk, const float* Wv,
                       const float* Wo, float* outDelta) {
    gemm(X,  Wq, nullptr, qb, SEQ, DKH, DM, DM, DKH, DKH, 0, 1.f, 0, NB, NH,
         (long long)SEQ * DM, 0, 0, WQS, QS1, QS2);
    gemm(KV, Wk, nullptr, kb, SEQ, DKH, DM, DM, DKH, DKH, 0, 1.f, 0, NB, NH,
         (long long)SEQ * DM, 0, 0, WQS, QS1, QS2);
    gemm(KV, Wv, nullptr, vb, SEQ, DKH, DM, DM, DKH, DKH, 0, 1.f, 0, NB, NH,
         (long long)SEQ * DM, 0, 0, WQS, QS1, QS2);
    // scores = Q @ K^T / sqrt(DK)   (transB: K stored [S, DK] N-major)
    gemm(qb, kb, nullptr, pb, SEQ, SEQ, DKH, DKH, DKH, SEQ, 1, 0.125f, 0, NB, NH,
         QS1, QS2, QS1, QS2, PS1, PS2);
    softmax512<<<dim3((unsigned)((size_t)NB * NH * SEQ / 8)), 256, 0, stream>>>(pb);
    // O = P @ V, written directly into head-concat [B, S, H*DK] layout
    gemm(pb, vb, nullptr, ocat, SEQ, DKH, SEQ, SEQ, DKH, NH * DKH, 0, 1.f, 0, NB, NH,
         PS1, PS2, QS1, QS2, (long long)SEQ * NH * DKH, (long long)DKH);
    // output projection, flat [B*S, D]
    gemm(ocat, Wo, nullptr, outDelta, NB * SEQ, DM, NH * DKH,
         NH * DKH, DM, DM, 0, 1.f, 0, 1, 1, 0, 0, 0, 0, 0, 0);
  };
  auto ffn = [&](float* x, const float* W1, const float* fb1,
                 const float* W2, const float* fb2) {
    gemm(x, W1, fb1, ffb, NB * SEQ, DFFN, DM, DM, DFFN, DFFN, 0, 1.f, 1,
         1, 1, 0, 0, 0, 0, 0, 0);
    gemm(ffb, W2, fb2, t1, NB * SEQ, DM, DFFN, DFFN, DM, DM, 0, 1.f, 0,
         1, 1, 0, 0, 0, 0, 0, 0);
  };

  const unsigned peBlocks = (unsigned)(ACT / 256);

  // ================= encoder =================
  add_pe<<<dim3(peBlocks), 256, 0, stream>>>(src, mem);
  for (int l = 0; l < LAYERS; ++l) {
    const size_t wOff = (size_t)l * NH * DM * DKH;
    attention(mem, mem, enc_Wq + wOff, enc_Wk + wOff, enc_Wv + wOff,
              enc_Wo + (size_t)l * NH * DKH * DM, t1);
    layernorm(mem, t1, enc_ln1g + l * DM, enc_ln1b + l * DM, mem);
    ffn(mem, enc_W1 + (size_t)l * DM * DFFN, enc_fb1 + (size_t)l * DFFN,
        enc_W2 + (size_t)l * DFFN * DM, enc_fb2 + (size_t)l * DM);
    layernorm(mem, t1, enc_ln2g + l * DM, enc_ln2b + l * DM, mem);
  }
  layernorm(mem, nullptr, fin_ln1g, fin_ln1b, mem);

  // ================= decoder =================
  add_pe<<<dim3(peBlocks), 256, 0, stream>>>(tgt, xbuf);
  for (int l = 0; l < LAYERS; ++l) {
    const size_t wOff = (size_t)l * NH * DM * DKH;
    // self-attention
    attention(xbuf, xbuf, dec_sWq + wOff, dec_sWk + wOff, dec_sWv + wOff,
              dec_sWo + (size_t)l * NH * DKH * DM, t1);
    layernorm(xbuf, t1, dec_ln1g + l * DM, dec_ln1b + l * DM, xbuf);
    // cross-attention (K/V from encoder memory)
    attention(xbuf, mem, dec_cWq + wOff, dec_cWk + wOff, dec_cWv + wOff,
              dec_cWo + (size_t)l * NH * DKH * DM, t1);
    layernorm(xbuf, t1, dec_ln2g + l * DM, dec_ln2b + l * DM, xbuf);
    // FFN
    ffn(xbuf, dec_W1 + (size_t)l * DM * DFFN, dec_fb1 + (size_t)l * DFFN,
        dec_W2 + (size_t)l * DFFN * DM, dec_fb2 + (size_t)l * DM);
    layernorm(xbuf, t1, dec_ln3g + l * DM, dec_ln3b + l * DM, xbuf);
  }
  layernorm(xbuf, nullptr, fin_ln2g, fin_ln2b, (float*)d_out);
}
